// PiNet2Model_64759516889489
// MI455X (gfx1250) — compile-verified, matrix operand order checked by reading, not run on hardware
//
#include <hip/hip_runtime.h>
#include <math.h>

// ---------------------------------------------------------------------------
// CDNA5 (gfx1250) WMMA helpers: wave32, D = A(16x32 bf16) x B(32x16 bf16) + C(f32)
// ---------------------------------------------------------------------------
typedef __attribute__((ext_vector_type(16))) __bf16 v16bf;
typedef __attribute__((ext_vector_type(8)))  float  v8f;
typedef __attribute__((ext_vector_type(4)))  unsigned int uint4v;

union FragU { v16bf v; uint4v q[2]; };

// A-fragment (and, symmetrically, B-fragment from a column-major weight buffer).
// ISA 7.12.2 16-bit A 16x32: lanes 0-15 rows M=0..15 hold K = 0..7 & 16..23,
// lanes 16-31 same rows hold K = 8..15 & 24..31. Element i of the v16bf maps to
// k = 8*hi + i (i<8) else 8 + 8*hi + i  -> two contiguous 8-element (16B) runs.
__device__ __forceinline__ v16bf load_frag_bf(const __bf16* base, int stride,
                                              int k0, int lane) {
  int m  = lane & 15;
  int hi = lane >> 4;
  const __bf16* row = base + m * stride + k0;
  FragU f;
  f.q[0] = *(const uint4v*)(row + 8 * hi);
  f.q[1] = *(const uint4v*)(row + 16 + 8 * hi);
  return f.v;
}

__device__ __forceinline__ v8f wmma_bf16(v16bf a, v16bf b, v8f c) {
  // (neg_a, A, neg_b, B, c_mod, C, reuse_a, reuse_b)
  return __builtin_amdgcn_wmma_f32_16x16x32_bf16(false, a, false, b,
                                                 (short)0, c, false, false);
}

// Branch-free tanh. gfx1250 has V_TANH_F32 (ISA lists TANH as a TRANS32 op);
// use the builtin when clang exposes it, else a no-branch exp2-based form
// (e = exp(-2|x|) in (0,1] -> no overflow path, no divergence).
__device__ __forceinline__ float fast_tanh(float x) {
#if __has_builtin(__builtin_amdgcn_tanhf)
  return __builtin_amdgcn_tanhf(x);
#else
  float e = __expf(-2.f * __builtin_fabsf(x));
  float t = (1.f - e) * __frcp_rn(1.f + e);
  return __builtin_copysignf(t, x);
#endif
}

// ---------------------------------------------------------------------------
// Init: p1 = embed[Z], p3 = 0, p3_bf = 0, out = 0   (re-run every launch:
// harness replays the graph and atomics accumulate into p1/p3/out)
// ---------------------------------------------------------------------------
__global__ void init_kernel(const int* __restrict__ Z,
                            const float* __restrict__ embed_w,
                            float* __restrict__ p1, float* __restrict__ p3,
                            __bf16* __restrict__ p3bf,
                            float* __restrict__ out, int N, int B) {
  int total = N * 192;
  for (int i = blockIdx.x * blockDim.x + threadIdx.x; i < total;
       i += gridDim.x * blockDim.x) {
    p3[i] = 0.f;
    p3bf[i] = (__bf16)0.f;
    if (i < N * 64) p1[i] = embed_w[Z[i >> 6] * 64 + (i & 63)];
    if (i < B) out[i] = 0.f;
  }
}

// Convert f32 weight W[K][N] -> bf16 transposed Wt[N][K] (column-major so the
// B-fragment loader gets contiguous 16B runs along K).
__global__ void cvt_tr_kernel(const float* __restrict__ src,
                              __bf16* __restrict__ dst, int K, int N) {
  int i = blockIdx.x * blockDim.x + threadIdx.x;
  if (i >= K * N) return;
  int k = i / N, n = i % N;
  dst[n * K + k] = (__bf16)src[i];
}

__global__ void snap_p3_kernel(const float* __restrict__ p3,
                               __bf16* __restrict__ p3bf, int n) {
  int i = blockIdx.x * blockDim.x + threadIdx.x;
  if (i < n) p3bf[i] = (__bf16)p3[i];
}

// ---------------------------------------------------------------------------
// Per-pair geometry: dist, unit direction, Gaussian RBF * cosine cutoff
// ---------------------------------------------------------------------------
__global__ void basis_kernel(const float* __restrict__ pd,
                             float* __restrict__ basis,
                             float* __restrict__ dhat, int P) {
  int p = blockIdx.x * blockDim.x + threadIdx.x;
  if (p >= P) return;
  float dx = pd[3 * p], dy = pd[3 * p + 1], dz = pd[3 * p + 2];
  float dist = __fsqrt_rn(dx * dx + dy * dy + dz * dz + 1e-12f);
  float inv = __frcp_rn(dist);
  dhat[3 * p] = dx * inv; dhat[3 * p + 1] = dy * inv; dhat[3 * p + 2] = dz * inv;
  float fc =
      (dist < 4.f) ? 0.5f * (__cosf(0.78539816339744831f * dist) + 1.f) : 0.f;
  const float sigma = 4.f / 15.f;
  const float isig = 15.f / 4.f;
#pragma unroll
  for (int b = 0; b < 16; ++b) {
    float t = (dist - sigma * (float)b) * isig;
    basis[16 * p + b] = __expf(-0.5f * t * t) * fc;
  }
}

// ---------------------------------------------------------------------------
// pp: h = tanh(tanh(p1 @ W1 + b1) @ W2 + b2)  -> bf16 h for the pair stage
// 128 threads = 4 waves; each wave owns one 16-col n-tile of a 16-atom tile.
// ---------------------------------------------------------------------------
__global__ __launch_bounds__(128) void pp_kernel(
    const float* __restrict__ p1, const __bf16* __restrict__ w1t,
    const float* __restrict__ b1, const __bf16* __restrict__ w2t,
    const float* __restrict__ b2, __bf16* __restrict__ h_out, int N) {
  __shared__ __align__(16) __bf16 sX[16][64];
  __shared__ __align__(16) __bf16 sH[16][64];
  int tid = threadIdx.x, lane = tid & 31, wave = tid >> 5;
  int a0 = blockIdx.x * 16;
  for (int e = tid; e < 1024; e += 128) {
    int m = e >> 6, k = e & 63;
    int a = a0 + m;
    sX[m][k] = (__bf16)((a < N) ? p1[a * 64 + k] : 0.f);
  }
  __syncthreads();
  int n0 = wave * 16;
  int n = lane & 15, hi = lane >> 4;
  v8f acc;
  { float bb = b1[n0 + n];
#pragma unroll
    for (int r = 0; r < 8; ++r) acc[r] = bb; }
#pragma unroll
  for (int ks = 0; ks < 2; ++ks)
    acc = wmma_bf16(load_frag_bf(&sX[0][0], 64, ks * 32, lane),
                    load_frag_bf(w1t + n0 * 64, 64, ks * 32, lane), acc);
#pragma unroll
  for (int r = 0; r < 8; ++r)
    sH[r + 8 * hi][n0 + n] = (__bf16)fast_tanh(acc[r]);
  __syncthreads();
  { float bb = b2[n0 + n];
#pragma unroll
    for (int r = 0; r < 8; ++r) acc[r] = bb; }
#pragma unroll
  for (int ks = 0; ks < 2; ++ks)
    acc = wmma_bf16(load_frag_bf(&sH[0][0], 64, ks * 32, lane),
                    load_frag_bf(w2t + n0 * 64, 64, ks * 32, lane), acc);
#pragma unroll
  for (int r = 0; r < 8; ++r) {
    int a = a0 + r + 8 * hi;
    if (a < N) h_out[a * 64 + n0 + n] = (__bf16)fast_tanh(acc[r]);
  }
}

// ---------------------------------------------------------------------------
// Fused pair pipeline for a 16-pair tile (256 threads = 8 waves):
//   gather h_i|h_j and p3_bf[j] -> LDS
//   g1 = tanh([16,128] @ pi_w1 + b1)                 (waves 0-3, WMMA)
//   g2 = tanh(g1 @ pi_w2 + b2)  [16,1024]            (8 waves x 8 tiles, WMMA)
//   i1[m][c] = sum_b g2[m][c*16+b]*basis[m][b]       (shfl_xor reduction:
//                                                     tile columns == rbf idx)
//   t  = tanh(i1 @ ii_w1);  u = t @ ii_w2 [16,192]   (WMMA)
//   p3j = p3_bf[j] @ pix_w  (3x [16,64]@[64,64])     (WMMA)
//   scatter: p1 += u[:, :64];  p3 += p3j*u[:,64:128] + dhat*u[:,128:192]
// All intermediates stay in LDS; g [P,64,16] (400 MB) is never materialized.
// ---------------------------------------------------------------------------
__global__ __launch_bounds__(256) void pair_kernel(
    const __bf16* __restrict__ h_bf, const __bf16* __restrict__ p3_bf,
    const float* __restrict__ basis, const float* __restrict__ dhat,
    const int* __restrict__ pair_i, const int* __restrict__ pair_j,
    const __bf16* __restrict__ piw1t, const float* __restrict__ pib1,
    const __bf16* __restrict__ piw2t, const float* __restrict__ pib2,
    const __bf16* __restrict__ iiw1t, const __bf16* __restrict__ iiw2t,
    const __bf16* __restrict__ pixwt,
    float* __restrict__ p1, float* __restrict__ p3, int P) {
  __shared__ __align__(16) __bf16 sA[16][128];    // [h_i | h_j]
  __shared__ __align__(16) __bf16 sP3[16][192];   // p3_bf[pair_j]
  __shared__ __align__(16) __bf16 sG1[16][64];
  __shared__ __align__(16) __bf16 sI1[16][64];
  __shared__ __align__(16) __bf16 sT[16][64];
  __shared__ float sBasis[16][16];
  __shared__ float sDhat[16][3];
  __shared__ float sII[16][192];
  __shared__ float sP3J[16][192];
  __shared__ int sPi[16], sPj[16];

  int tid = threadIdx.x, lane = tid & 31, wave = tid >> 5;
  int p0 = blockIdx.x * 16;
  int n = lane & 15, hi = lane >> 4;

  if (tid < 16) {
    int p = p0 + tid;
    bool ok = p < P;
    sPi[tid] = ok ? pair_i[p] : 0;
    sPj[tid] = ok ? pair_j[p] : 0;
  }
  if (tid >= 64 && tid < 112) {
    int e = tid - 64; int m = e / 3, x = e % 3;
    sDhat[m][x] = (p0 + m < P) ? dhat[(p0 + m) * 3 + x] : 0.f;
  }
  {
    int m = tid >> 4, b = tid & 15;
    sBasis[m][b] = (p0 + m < P) ? basis[(p0 + m) * 16 + b] : 0.f;
  }
  __syncthreads();

  for (int e = tid; e < 2048; e += 256) {
    int m = e >> 7, k = e & 127;
    sA[m][k] = (k < 64) ? h_bf[sPi[m] * 64 + k] : h_bf[sPj[m] * 64 + (k - 64)];
  }
  for (int e = tid; e < 3072; e += 256) {
    int m = e / 192, k = e - m * 192;
    sP3[m][k] = p3_bf[sPj[m] * 192 + k];
  }
  __syncthreads();

  // ---- pi layer 1: [16,128] @ [128,64] ----
  if (wave < 4) {
    int n0 = wave * 16;
    v8f acc;
    float bb = pib1[n0 + n];
#pragma unroll
    for (int r = 0; r < 8; ++r) acc[r] = bb;
#pragma unroll
    for (int ks = 0; ks < 4; ++ks)
      acc = wmma_bf16(load_frag_bf(&sA[0][0], 128, ks * 32, lane),
                      load_frag_bf(piw1t + n0 * 128, 128, ks * 32, lane), acc);
#pragma unroll
    for (int r = 0; r < 8; ++r)
      sG1[r + 8 * hi][n0 + n] = (__bf16)fast_tanh(acc[r]);
  }
  __syncthreads();

  // ---- pi layer 2 + basis contraction: wave owns channels 8w..8w+7 ----
  {
    v8f acc[8];
#pragma unroll
    for (int t = 0; t < 8; ++t) {
      float bb = pib2[(wave * 8 + t) * 16 + n];
#pragma unroll
      for (int r = 0; r < 8; ++r) acc[t][r] = bb;
    }
#pragma unroll
    for (int ks = 0; ks < 2; ++ks) {
      v16bf a = load_frag_bf(&sG1[0][0], 64, ks * 32, lane);
#pragma unroll
      for (int t = 0; t < 8; ++t)
        acc[t] = wmma_bf16(
            a, load_frag_bf(piw2t + (wave * 8 + t) * 16 * 64, 64, ks * 32, lane),
            acc[t]);
    }
    // column n within a tile IS the rbf index b; this lane needs basis rows
    // m = r + 8*hi at column n -- hoist the 8 LDS reads out of the tile loop.
    float bas[8];
#pragma unroll
    for (int r = 0; r < 8; ++r) bas[r] = sBasis[r + 8 * hi][n];
#pragma unroll
    for (int t = 0; t < 8; ++t) {
      int c = wave * 8 + t;
#pragma unroll
      for (int r = 0; r < 8; ++r) {
        int m = r + 8 * hi;
        float v = fast_tanh(acc[t][r]) * bas[r];
        v += __shfl_xor(v, 1, 32);
        v += __shfl_xor(v, 2, 32);
        v += __shfl_xor(v, 4, 32);
        v += __shfl_xor(v, 8, 32);
        if (n == 0) sI1[m][c] = (__bf16)v;   // lanes 0 & 16 write m=0..7 / 8..15
      }
    }
  }
  __syncthreads();

  // ---- ii layer 1: t = tanh(i1 @ ii_w1) (bias-free) ----
  if (wave < 4) {
    int n0 = wave * 16;
    v8f acc;
#pragma unroll
    for (int r = 0; r < 8; ++r) acc[r] = 0.f;
#pragma unroll
    for (int ks = 0; ks < 2; ++ks)
      acc = wmma_bf16(load_frag_bf(&sI1[0][0], 64, ks * 32, lane),
                      load_frag_bf(iiw1t + n0 * 64, 64, ks * 32, lane), acc);
#pragma unroll
    for (int r = 0; r < 8; ++r)
      sT[r + 8 * hi][n0 + n] = (__bf16)fast_tanh(acc[r]);
  }
  __syncthreads();

  // ---- ii layer 2 (12 tiles) + pix (3 x 4 tiles): 24 jobs over 8 waves ----
#pragma unroll
  for (int jj = 0; jj < 3; ++jj) {
    int job = wave * 3 + jj;
    v8f acc;
#pragma unroll
    for (int r = 0; r < 8; ++r) acc[r] = 0.f;
    if (job < 12) {
      int n0 = job * 16;
#pragma unroll
      for (int ks = 0; ks < 2; ++ks)
        acc = wmma_bf16(load_frag_bf(&sT[0][0], 64, ks * 32, lane),
                        load_frag_bf(iiw2t + n0 * 64, 64, ks * 32, lane), acc);
#pragma unroll
      for (int r = 0; r < 8; ++r) sII[r + 8 * hi][n0 + n] = acc[r];
    } else {
      int q = job - 12; int x = q >> 2; int n0 = (q & 3) * 16;
#pragma unroll
      for (int ks = 0; ks < 2; ++ks)
        acc = wmma_bf16(load_frag_bf(&sP3[0][x * 64], 192, ks * 32, lane),
                        load_frag_bf(pixwt + n0 * 64, 64, ks * 32, lane), acc);
#pragma unroll
      for (int r = 0; r < 8; ++r) sP3J[r + 8 * hi][x * 64 + n0 + n] = acc[r];
    }
  }
  __syncthreads();

  // ---- scatter-sum back to atoms (fp32 global atomics) ----
  for (int e = tid; e < 1024; e += 256) {
    int m = e >> 6, c = e & 63;
    if (p0 + m < P) atomicAdd(&p1[sPi[m] * 64 + c], sII[m][c]);
  }
  for (int e = tid; e < 3072; e += 256) {
    int m = e / 192, k = e - m * 192;
    if (p0 + m < P) {
      int x = k >> 6, c = k & 63;
      float v = sP3J[m][k] * sII[m][64 + c] + sDhat[m][x] * sII[m][128 + c];
      atomicAdd(&p3[sPi[m] * 192 + k], v);
    }
  }
}

// ---------------------------------------------------------------------------
// Readout: atom_e = tanh(p1 @ ro_w1 + b1) @ ro_w2 + b2 ; segment-sum to out
// ---------------------------------------------------------------------------
__global__ void readout_kernel(const float* __restrict__ p1,
                               const float* __restrict__ w1,
                               const float* __restrict__ b1,
                               const float* __restrict__ w2,
                               const float* __restrict__ b2,
                               const int* __restrict__ atom_batch,
                               float* __restrict__ out, int N) {
  int a = blockIdx.x * blockDim.x + threadIdx.x;
  if (a >= N) return;
  const float* row = p1 + a * 64;
  float e = b2[0];
#pragma unroll 4
  for (int j = 0; j < 32; ++j) {
    float s = b1[j];
    for (int k = 0; k < 64; ++k) s += row[k] * w1[k * 32 + j];
    e += fast_tanh(s) * w2[j];
  }
  atomicAdd(&out[atom_batch[a]], e);
}

// ---------------------------------------------------------------------------
extern "C" void kernel_launch(void* const* d_in, const int* in_sizes, int n_in,
                              void* d_out, int out_size, void* d_ws,
                              size_t ws_size, hipStream_t stream) {
  const int*   Z         = (const int*)d_in[0];
  const float* pair_diff = (const float*)d_in[1];
  const int*   pair_i    = (const int*)d_in[2];
  const int*   pair_j    = (const int*)d_in[3];
  const int*   atom_batch= (const int*)d_in[4];
  const float* embed_w   = (const float*)d_in[6];
  const float* pp_w1 = (const float*)d_in[7];
  const float* pp_b1 = (const float*)d_in[8];
  const float* pp_w2 = (const float*)d_in[9];
  const float* pp_b2 = (const float*)d_in[10];
  const float* pi_w1 = (const float*)d_in[11];
  const float* pi_b1 = (const float*)d_in[12];
  const float* pi_w2 = (const float*)d_in[13];
  const float* pi_b2 = (const float*)d_in[14];
  const float* ii_w1 = (const float*)d_in[15];
  const float* ii_w2 = (const float*)d_in[16];
  const float* pix_w = (const float*)d_in[17];
  const float* ro_w1 = (const float*)d_in[18];
  const float* ro_b1 = (const float*)d_in[19];
  const float* ro_w2 = (const float*)d_in[20];
  const float* ro_b2 = (const float*)d_in[21];
  (void)n_in; (void)ws_size;

  int N = in_sizes[0];
  int P = in_sizes[2];
  int B = out_size;
  int DEPTH = in_sizes[7] / (64 * 64);
  float* out = (float*)d_out;

  char* ws = (char*)d_ws;
  size_t off = 0;
  auto take = [&](size_t bytes) {
    char* p = ws + off;
    off = (off + bytes + 255) & ~(size_t)255;
    return p;
  };
  float*  p1    = (float*)take((size_t)N * 64 * 4);
  float*  p3    = (float*)take((size_t)N * 192 * 4);
  __bf16* hbf   = (__bf16*)take((size_t)N * 64 * 2);
  __bf16* p3bf  = (__bf16*)take((size_t)N * 192 * 2);
  float*  basis = (float*)take((size_t)P * 16 * 4);
  float*  dhat  = (float*)take((size_t)P * 3 * 4);
  __bf16* ppw1t = (__bf16*)take((size_t)DEPTH * 64 * 64 * 2);
  __bf16* ppw2t = (__bf16*)take((size_t)DEPTH * 64 * 64 * 2);
  __bf16* piw1t = (__bf16*)take((size_t)DEPTH * 128 * 64 * 2);
  __bf16* piw2t = (__bf16*)take((size_t)DEPTH * 64 * 1024 * 2);
  __bf16* iiw1t = (__bf16*)take((size_t)DEPTH * 64 * 64 * 2);
  __bf16* iiw2t = (__bf16*)take((size_t)DEPTH * 64 * 192 * 2);
  __bf16* pixwt = (__bf16*)take((size_t)DEPTH * 64 * 64 * 2);

  init_kernel<<<512, 256, 0, stream>>>(Z, embed_w, p1, p3, p3bf, out, N, B);
  basis_kernel<<<(P + 255) / 256, 256, 0, stream>>>(pair_diff, basis, dhat, P);

  for (int d = 0; d < DEPTH; ++d) {
    cvt_tr_kernel<<<(4096 + 255) / 256, 256, 0, stream>>>(pp_w1 + d * 4096, ppw1t + d * 4096, 64, 64);
    cvt_tr_kernel<<<(4096 + 255) / 256, 256, 0, stream>>>(pp_w2 + d * 4096, ppw2t + d * 4096, 64, 64);
    cvt_tr_kernel<<<(8192 + 255) / 256, 256, 0, stream>>>(pi_w1 + d * 8192, piw1t + d * 8192, 128, 64);
    cvt_tr_kernel<<<(65536 + 255) / 256, 256, 0, stream>>>(pi_w2 + d * 65536, piw2t + d * 65536, 64, 1024);
    cvt_tr_kernel<<<(4096 + 255) / 256, 256, 0, stream>>>(ii_w1 + d * 4096, iiw1t + d * 4096, 64, 64);
    cvt_tr_kernel<<<(12288 + 255) / 256, 256, 0, stream>>>(ii_w2 + d * 12288, iiw2t + d * 12288, 64, 192);
    cvt_tr_kernel<<<(4096 + 255) / 256, 256, 0, stream>>>(pix_w + d * 4096, pixwt + d * 4096, 64, 64);
  }

  for (int d = 0; d < DEPTH; ++d) {
    pp_kernel<<<(N + 15) / 16, 128, 0, stream>>>(
        p1, ppw1t + d * 4096, pp_b1 + d * 64, ppw2t + d * 4096, pp_b2 + d * 64,
        hbf, N);
    snap_p3_kernel<<<(N * 192 + 255) / 256, 256, 0, stream>>>(p3, p3bf, N * 192);
    pair_kernel<<<(P + 15) / 16, 256, 0, stream>>>(
        hbf, p3bf, basis, dhat, pair_i, pair_j,
        piw1t + d * 8192, pi_b1 + d * 64, piw2t + d * 65536, pi_b2 + d * 1024,
        iiw1t + d * 4096, iiw2t + d * 12288, pixwt + d * 4096, p1, p3, P);
  }

  readout_kernel<<<(N + 127) / 128, 128, 0, stream>>>(
      p1, ro_w1, ro_b1, ro_w2, ro_b2, atom_batch, out, N);
}
